// MixedDecoder_74423193305292
// MI455X (gfx1250) — compile-verified, matrix operand order checked by reading, not run on hardware
//
#include <hip/hip_runtime.h>

// ---- problem constants ----
#define B_SZ  4096
#define LAT   64
#define FRAME 512
#define HID   1024
#define NE    8
#define GH    128
#define IN1   576     // LAT + FRAME
#define IN2   1088    // LAT + HID
#define OUTF  512

typedef unsigned short us;
typedef unsigned int   u32;
typedef __attribute__((ext_vector_type(4)))  u32    u32x4;   // trivially-constructible 16B vector
typedef __attribute__((ext_vector_type(16))) __bf16 v16bf;
typedef __attribute__((ext_vector_type(8)))  float  v8f;

__device__ __forceinline__ us f2bf(float f) {
  union { float f; u32 u; } x; x.f = f;
  u32 u = x.u;
  u32 r = (u + 0x7fffu + ((u >> 16) & 1u)) >> 16;  // round-to-nearest-even
  return (us)r;
}
__device__ __forceinline__ float eluf(float x) {
  return x > 0.0f ? x : (__expf(x) - 1.0f);
}

// CDNA5 async global->LDS copy (ASYNCcnt), 16 bytes per lane.
__device__ __forceinline__ void async16(u32 lds_addr, const us* gaddr) {
  asm volatile("global_load_async_to_lds_b128 %0, %1, off"
               :: "v"(lds_addr), "v"(gaddr) : "memory");
}
__device__ __forceinline__ void wait_async0() {
  asm volatile("s_wait_asynccnt 0x0" ::: "memory");
}

// ---------------------------------------------------------------------------
// Pack z|c into bf16 activation buffers; z also seeds x1/x2 (ld = IN2).
// ---------------------------------------------------------------------------
__global__ __launch_bounds__(256) void pack_x(const float* __restrict__ z,
                                              const float* __restrict__ c,
                                              us* __restrict__ x0,
                                              us* __restrict__ x1,
                                              us* __restrict__ x2) {
  int id = blockIdx.x * 256 + threadIdx.x;
  if (id >= B_SZ * IN1) return;
  int b = id / IN1, i = id % IN1;
  if (i < LAT) {
    us v = f2bf(z[b * LAT + i]);
    x0[(size_t)b * IN1 + i] = v;
    x1[(size_t)b * IN2 + i] = v;
    x2[(size_t)b * IN2 + i] = v;
  } else {
    x0[(size_t)b * IN1 + i] = f2bf(c[b * FRAME + (i - LAT)]);
  }
}

// ---------------------------------------------------------------------------
// Gate MLP (f32): 4 rows per 128-thread block -> softmax coefficients [B, 8].
// ---------------------------------------------------------------------------
#define GROWS 4
__global__ __launch_bounds__(128) void gate_kernel(
    const float* __restrict__ z,  const float* __restrict__ c,
    const float* __restrict__ gw1, const float* __restrict__ gb1,
    const float* __restrict__ gw2, const float* __restrict__ gb2,
    const float* __restrict__ gw3, const float* __restrict__ gb3,
    float* __restrict__ coef) {
  __shared__ float xs[GROWS * IN1];
  __shared__ float h1[GROWS * GH];
  __shared__ float h2[GROWS * GH];
  __shared__ float lg[GROWS * NE];
  const int tid = threadIdx.x;
  const int r0  = blockIdx.x * GROWS;

  for (int idx = tid; idx < GROWS * IN1; idx += 128) {
    int rr = idx / IN1, i = idx % IN1, b = r0 + rr;
    xs[idx] = (i < LAT) ? z[b * LAT + i] : c[b * FRAME + (i - LAT)];
  }
  __syncthreads();
  {
    float acc[GROWS];
    float bb = gb1[tid];
#pragma unroll
    for (int rr = 0; rr < GROWS; rr++) acc[rr] = bb;
    for (int i = 0; i < IN1; i++) {
      float w = gw1[i * GH + tid];
#pragma unroll
      for (int rr = 0; rr < GROWS; rr++) acc[rr] = fmaf(xs[rr * IN1 + i], w, acc[rr]);
    }
#pragma unroll
    for (int rr = 0; rr < GROWS; rr++) h1[rr * GH + tid] = eluf(acc[rr]);
  }
  __syncthreads();
  {
    float acc[GROWS];
    float bb = gb2[tid];
#pragma unroll
    for (int rr = 0; rr < GROWS; rr++) acc[rr] = bb;
    for (int i = 0; i < GH; i++) {
      float w = gw2[i * GH + tid];
#pragma unroll
      for (int rr = 0; rr < GROWS; rr++) acc[rr] = fmaf(h1[rr * GH + i], w, acc[rr]);
    }
#pragma unroll
    for (int rr = 0; rr < GROWS; rr++) h2[rr * GH + tid] = eluf(acc[rr]);
  }
  __syncthreads();
  if (tid < GROWS * NE) {
    int rr = tid >> 3, e = tid & 7;
    float acc = gb3[e];
    for (int i = 0; i < GH; i++) acc = fmaf(h2[rr * GH + i], gw3[i * NE + e], acc);
    lg[rr * NE + e] = acc;
  }
  __syncthreads();
  if (tid < GROWS) {
    int rr = tid;
    float m = lg[rr * NE];
#pragma unroll
    for (int e = 1; e < NE; e++) m = fmaxf(m, lg[rr * NE + e]);
    float ex[NE], s = 0.0f;
#pragma unroll
    for (int e = 0; e < NE; e++) { ex[e] = __expf(lg[rr * NE + e] - m); s += ex[e]; }
    float inv = 1.0f / s;
#pragma unroll
    for (int e = 0; e < NE; e++) coef[(size_t)(r0 + rr) * NE + e] = ex[e] * inv;
  }
}

// ---------------------------------------------------------------------------
// Transpose + f32->bf16: w[e][IN][OUT] -> wt[e][OUT][IN]  (all dims % 32 == 0)
// ---------------------------------------------------------------------------
__global__ __launch_bounds__(256) void wcvt_t(const float* __restrict__ w,
                                              us* __restrict__ wt,
                                              int INDIM, int OUTD) {
  __shared__ float tile[32][33];
  const int e  = blockIdx.z;
  const float* src = w  + (size_t)e * INDIM * OUTD;
  us*          dst = wt + (size_t)e * INDIM * OUTD;
  const int o0 = blockIdx.x * 32, i0 = blockIdx.y * 32;
  const int tx = threadIdx.x & 31, ty = threadIdx.x >> 5;
  for (int r = ty; r < 32; r += 8)
    tile[r][tx] = src[(size_t)(i0 + r) * OUTD + o0 + tx];
  __syncthreads();
  for (int r = ty; r < 32; r += 8)
    dst[(size_t)(o0 + r) * INDIM + i0 + tx] = f2bf(tile[tx][r]);
}

// ---------------------------------------------------------------------------
// Fused MoE GEMM:  out[b,n] = sum_e coef[b,e]*(X[b,:] @ We[:,n]) + coef @ bias
// X: bf16 [B, INDIM], Wt: bf16 [E][OUTD][INDIM], coef: f32 [B,8]
// Workgroup tile 128(M) x 128(N); 8 waves as 4(M) x 2(N), each wave 32x64.
// Flattened K loop over NE*INDIM; tiles staged with GLOBAL_LOAD_ASYNC_TO_LDS
// (double-buffered, ASYNCcnt-tracked, issued after the barrier so they overlap
// the whole WMMA phase). All six fragments are loaded into distinct registers
// before the 8-WMMA block so ds_load waits stagger instead of serializing.
// MODE 0: ELU -> bf16 into next activation buf at column offset LAT (ld IN2)
// MODE 1: f32 into d_out [B, OUTF]
// ---------------------------------------------------------------------------
union Frag { u32x4 q[2]; v16bf v; };

template <int INDIM, int OUTD, int MODE>
__global__ __launch_bounds__(256) void moe_gemm(const us* __restrict__ X,
                                                const us* __restrict__ Wt,
                                                const float* __restrict__ coef,
                                                const float* __restrict__ bias,
                                                void* __restrict__ dst) {
  __shared__ __align__(16) us    lsA[2][128 * 32];   // 2 x 8 KB
  __shared__ __align__(16) us    lsB[2][128 * 32];   // 2 x 8 KB
  __shared__ __align__(16) float lsC[128 * NE];      // 4 KB coef block

  const int tid  = threadIdx.x;
  const int lane = tid & 31;
  const int wave = tid >> 5;
  const int wm   = wave & 3;        // 4 waves along M, 32 rows each
  const int wn   = wave >> 2;       // 2 waves along N, 64 cols each
  const int mblk = blockIdx.x * 128;
  const int nblk = blockIdx.y * 128;

  const int srow = tid >> 1;              // staging row (0..127)
  const int scol = (tid & 1) * 16;        // staging col base; chunks at +0, +8

  const int c0   = (lane & 16) ? 8 : 0;   // ISA 16-bit K-chunk base
  const int fra  = wm * 32 + (lane & 15); // A-frag row, mt=0 (mt=1: +16)
  const int hi8  = (lane & 16) ? 8 : 0;

  constexpr int NSTEP = NE * INDIM / 32;
  static_assert(NSTEP % 2 == 0, "even step count required for buffer reuse");
  const v8f vzero = {0.f,0.f,0.f,0.f,0.f,0.f,0.f,0.f};

  // stage this workgroup's 128x8 coef block into LDS (one 16B load/thread)
  ((u32x4*)lsC)[tid] = ((const u32x4*)(coef + (size_t)mblk * NE))[tid];

  v8f acc[2][4], acce[2][4];
#pragma unroll
  for (int mt = 0; mt < 2; mt++)
#pragma unroll
    for (int t = 0; t < 4; t++) { acc[mt][t] = vzero; acce[mt][t] = vzero; }

  // per-thread LDS byte addresses for async staging (low 32 bits = LDS offset)
  const u32 ldsA0 = (u32)(uintptr_t)&lsA[0][srow * 32 + scol];
  const u32 ldsB0 = (u32)(uintptr_t)&lsB[0][srow * 32 + scol];
  const us* Xrow  = X + (size_t)(mblk + srow) * INDIM + scol;

  auto stage = [&](int step) {
    const int k0 = step * 32;
    const int e  = k0 / INDIM;
    const int kk = k0 - e * INDIM;
    const u32 boff = (u32)(step & 1) * (u32)(128 * 32 * sizeof(us));
    const us* gA = Xrow + kk;
    const us* gB = Wt + (size_t)(e * OUTD + nblk + srow) * INDIM + kk + scol;
    async16(ldsA0 + boff,      gA);
    async16(ldsA0 + boff + 16, gA + 8);
    async16(ldsB0 + boff,      gB);
    async16(ldsB0 + boff + 16, gB + 8);
  };

  stage(0);   // prologue

  for (int step = 0; step < NSTEP; ++step) {
    wait_async0();        // my share of tile `step` has landed in LDS
    __syncthreads();      // everyone's landed; tile step-1 readers all done
    if (step + 1 < NSTEP) stage(step + 1);   // overlaps the compute below

    const us* bA = lsA[step & 1];
    const us* bB = lsB[step & 1];
    // load all fragments into distinct registers: waits stagger, no WAR chain
    Frag fa0, fa1, fb[4];
    fa0.q[0] = *(const u32x4*)&bA[fra * 32 + c0];
    fa0.q[1] = *(const u32x4*)&bA[fra * 32 + c0 + 16];
    fa1.q[0] = *(const u32x4*)&bA[(fra + 16) * 32 + c0];
    fa1.q[1] = *(const u32x4*)&bA[(fra + 16) * 32 + c0 + 16];
#pragma unroll
    for (int t = 0; t < 4; t++) {
      const int nrow = wn * 64 + t * 16 + (lane & 15);
      fb[t].q[0] = *(const u32x4*)&bB[nrow * 32 + c0];
      fb[t].q[1] = *(const u32x4*)&bB[nrow * 32 + c0 + 16];
    }
#pragma unroll
    for (int t = 0; t < 4; t++) {
      acce[0][t] = __builtin_amdgcn_wmma_f32_16x16x32_bf16(
          false, fa0.v, false, fb[t].v, (short)0, acce[0][t], false, false);
      acce[1][t] = __builtin_amdgcn_wmma_f32_16x16x32_bf16(
          false, fa1.v, false, fb[t].v, (short)0, acce[1][t], false, false);
    }

    // expert boundary: fold partials with per-row gate coefficient (from LDS)
    if (((step + 1) * 32) % INDIM == 0) {
      const int e = (step * 32) / INDIM;
#pragma unroll
      for (int mt = 0; mt < 2; mt++) {
        const int rl = wm * 32 + mt * 16 + hi8;   // local row base
        v8f cv;
#pragma unroll
        for (int v = 0; v < 8; v++) cv[v] = lsC[(rl + v) * NE + e];
#pragma unroll
        for (int t = 0; t < 4; t++) {
          acc[mt][t] += cv * acce[mt][t];
          acce[mt][t] = vzero;
        }
      }
    }
  }

  // ---- bias as one extra WMMA K-step: A = coef (bf16), B = bias^T ----
  __syncthreads();   // last compute step finished reading its buffer
  if (tid < 128) {
#pragma unroll
    for (int e = 0; e < NE; e++) lsA[0][tid * 32 + e] = f2bf(lsC[tid * NE + e]);
    for (int k = NE; k < 32; k++) lsA[0][tid * 32 + k] = 0;
    const int n = nblk + tid;
#pragma unroll
    for (int e = 0; e < NE; e++) lsB[0][tid * 32 + e] = f2bf(bias[e * OUTD + n]);
    for (int k = NE; k < 32; k++) lsB[0][tid * 32 + k] = 0;
  }
  __syncthreads();
  {
    Frag fa0, fa1, fb[4];
    fa0.q[0] = *(const u32x4*)&lsA[0][fra * 32 + c0];
    fa0.q[1] = *(const u32x4*)&lsA[0][fra * 32 + c0 + 16];
    fa1.q[0] = *(const u32x4*)&lsA[0][(fra + 16) * 32 + c0];
    fa1.q[1] = *(const u32x4*)&lsA[0][(fra + 16) * 32 + c0 + 16];
#pragma unroll
    for (int t = 0; t < 4; t++) {
      const int nrow = wn * 64 + t * 16 + (lane & 15);
      fb[t].q[0] = *(const u32x4*)&lsB[0][nrow * 32 + c0];
      fb[t].q[1] = *(const u32x4*)&lsB[0][nrow * 32 + c0 + 16];
    }
#pragma unroll
    for (int t = 0; t < 4; t++) {
      acc[0][t] = __builtin_amdgcn_wmma_f32_16x16x32_bf16(
          false, fa0.v, false, fb[t].v, (short)0, acc[0][t], false, false);
      acc[1][t] = __builtin_amdgcn_wmma_f32_16x16x32_bf16(
          false, fa1.v, false, fb[t].v, (short)0, acc[1][t], false, false);
    }
  }

  // ---- epilogue ----
  const int cb = nblk + wn * 64 + (lane & 15);
#pragma unroll
  for (int mt = 0; mt < 2; mt++) {
    const int rbase = mblk + wm * 32 + mt * 16 + hi8;
#pragma unroll
    for (int t = 0; t < 4; t++) {
#pragma unroll
      for (int v = 0; v < 8; v++) {
        const float val = acc[mt][t][v];
        const int r = rbase + v;
        const int col = cb + t * 16;
        if (MODE == 0) {
          ((us*)dst)[(size_t)r * IN2 + LAT + col] = f2bf(eluf(val));
        } else {
          ((float*)dst)[(size_t)r * OUTF + col] = val;
        }
      }
    }
  }
}

// ---------------------------------------------------------------------------
// workspace layout (bytes)
// ---------------------------------------------------------------------------
#define OFF_COEF 0u
#define OFF_X0   131072u                       // B*8*4
#define OFF_X1   (OFF_X0 + 4718592u)           // B*IN1*2
#define OFF_X2   (OFF_X1 + 8912896u)           // B*IN2*2
#define OFF_W0T  (OFF_X2 + 8912896u)
#define OFF_W1T  (OFF_W0T + 9437184u)          // 8*576*1024*2
#define OFF_W2T  (OFF_W1T + 17825792u)         // 8*1088*1024*2
// total = OFF_W2T + 8912896 = ~58.9 MB

extern "C" void kernel_launch(void* const* d_in, const int* in_sizes, int n_in,
                              void* d_out, int out_size, void* d_ws, size_t ws_size,
                              hipStream_t stream) {
  const float* z   = (const float*)d_in[0];
  const float* c   = (const float*)d_in[1];
  const float* w0  = (const float*)d_in[2];
  const float* b0  = (const float*)d_in[3];
  const float* w1  = (const float*)d_in[4];
  const float* b1  = (const float*)d_in[5];
  const float* w2  = (const float*)d_in[6];
  const float* b2  = (const float*)d_in[7];
  const float* gw1 = (const float*)d_in[8];
  const float* gb1 = (const float*)d_in[9];
  const float* gw2 = (const float*)d_in[10];
  const float* gb2 = (const float*)d_in[11];
  const float* gw3 = (const float*)d_in[12];
  const float* gb3 = (const float*)d_in[13];

  char* ws = (char*)d_ws;
  float* coef = (float*)(ws + OFF_COEF);
  us* x0  = (us*)(ws + OFF_X0);
  us* x1  = (us*)(ws + OFF_X1);
  us* x2  = (us*)(ws + OFF_X2);
  us* w0t = (us*)(ws + OFF_W0T);
  us* w1t = (us*)(ws + OFF_W1T);
  us* w2t = (us*)(ws + OFF_W2T);

  // independent prep
  pack_x<<<(B_SZ * IN1 + 255) / 256, 256, 0, stream>>>(z, c, x0, x1, x2);
  gate_kernel<<<B_SZ / GROWS, 128, 0, stream>>>(z, c, gw1, gb1, gw2, gb2, gw3, gb3, coef);
  wcvt_t<<<dim3(HID / 32, IN1 / 32, NE), 256, 0, stream>>>(w0, w0t, IN1, HID);
  wcvt_t<<<dim3(HID / 32, IN2 / 32, NE), 256, 0, stream>>>(w1, w1t, IN2, HID);
  wcvt_t<<<dim3(OUTF / 32, IN2 / 32, NE), 256, 0, stream>>>(w2, w2t, IN2, OUTF);

  // expert stack
  moe_gemm<IN1, HID, 0><<<dim3(B_SZ / 128, HID / 128), 256, 0, stream>>>(
      x0, w0t, coef, b0, (void*)x1);
  moe_gemm<IN2, HID, 0><<<dim3(B_SZ / 128, HID / 128), 256, 0, stream>>>(
      x1, w1t, coef, b1, (void*)x2);
  moe_gemm<IN2, OUTF, 1><<<dim3(B_SZ / 128, OUTF / 128), 256, 0, stream>>>(
      x2, w2t, coef, b2, d_out);
}
